// VectorizedSpikeGraph_25451976196440
// MI455X (gfx1250) — matrix operand order repaired and checked by reading, not compiled, not run on hardware
//
#include <hip/hip_runtime.h>

// ---------------------------------------------------------------------------
// LIF spiking network on MI455X (gfx1250, wave32, WMMA).
//   I_ext = X @ W_in^T + b        (bf16 WMMA, K=1024, one launch)
//   repeat 64x:
//     I = I_ext + s @ W_eff^T     (bf16 WMMA, K=2048)
//     V = 0.9 V + I; s' = (V>=1); V *= (1-s'); rate += s'/64
// Spikes are exactly representable in bf16; only W is quantized (2^-9 rel).
// W_eff (8 MB bf16) stays resident in the 192 MB L2 across the 64 steps, so
// we use many small workgroups (128 WGs x 4 waves) to spread each serialized
// time step across as many WGPs as possible; redundant A/B reads hit L2.
// ---------------------------------------------------------------------------

typedef __attribute__((ext_vector_type(16))) __bf16 v16bf;
typedef __attribute__((ext_vector_type(8)))  __bf16 v8bf;
typedef __attribute__((ext_vector_type(8)))  float  v8f;

#define BATCH  256
#define DIN    1024
#define NN     2048
#define TSTEPS 64
#define DECAY  0.9f
#define VTH    1.0f

// ---------------- prep / elementwise kernels ----------------

__global__ void k_maskmul_bf16(const float* __restrict__ a,
                               const float* __restrict__ m,
                               __bf16* __restrict__ o, int n) {
  int i = blockIdx.x * blockDim.x + threadIdx.x;
  int stride = gridDim.x * blockDim.x;
  for (; i < n; i += stride) o[i] = (__bf16)(a[i] * m[i]);
}

__global__ void k_cvt_bf16(const float* __restrict__ a,
                           __bf16* __restrict__ o, int n) {
  int i = blockIdx.x * blockDim.x + threadIdx.x;
  int stride = gridDim.x * blockDim.x;
  for (; i < n; i += stride) o[i] = (__bf16)a[i];
}

__global__ void k_zero_f32(float* __restrict__ o, int n) {
  int i = blockIdx.x * blockDim.x + threadIdx.x;
  int stride = gridDim.x * blockDim.x;
  for (; i < n; i += stride) o[i] = 0.0f;
}

__global__ void k_zero_bf16(__bf16* __restrict__ o, int n) {
  int i = blockIdx.x * blockDim.x + threadIdx.x;
  int stride = gridDim.x * blockDim.x;
  for (; i < n; i += stride) o[i] = (__bf16)0.0f;
}

// ---------------- WMMA tile GEMM core ----------------
// Computes a 32x32 accumulator tile per wave (2x2 subtiles of 16x16) for
// C[m,n] = sum_k A[m,k] * Bw[n,k]   (i.e. A @ Bw^T), A/Bw row-major bf16.
//
// ISA lane layouts (wave32):
//  A 16x32 bf16 : lane L holds row m=L&15; K chunks [8*(L>>4)..+7] in v0..v3
//                 and [16+8*(L>>4)..+7] in v4..v7  -> two b128 loads.
//  B 32x16 bf16 : lane L holds col n=L&15; K chunks [8*(L>>4)..+7] in v0..v3
//                 and [16+8*(L>>4)..+7] in v4..v7  -> two b128 loads.
__device__ __forceinline__ void wmma_tile_gemm(
    const __bf16* __restrict__ A, int lda,
    const __bf16* __restrict__ Bw, int ldb,
    int K, int mbase, int nbase, int lane, v8f acc[2][2]) {
  const int l15 = lane & 15;
  const int kh  = lane >> 4;  // 0 or 1
  for (int k0 = 0; k0 < K; k0 += 32) {
    v16bf a[2], b[2];
#pragma unroll
    for (int sm = 0; sm < 2; ++sm) {
      const __bf16* ap = A + (size_t)(mbase + sm * 16 + l15) * lda + k0;
      v8bf lo = *(const v8bf*)(ap + kh * 8);
      v8bf hi = *(const v8bf*)(ap + 16 + kh * 8);
#pragma unroll
      for (int i = 0; i < 8; ++i) { a[sm][i] = lo[i]; a[sm][i + 8] = hi[i]; }
    }
#pragma unroll
    for (int sn = 0; sn < 2; ++sn) {
      const __bf16* bp = Bw + (size_t)(nbase + sn * 16 + l15) * ldb + k0 + kh * 8;
      v8bf lo = *(const v8bf*)(bp);
      v8bf hi = *(const v8bf*)(bp + 16);
#pragma unroll
      for (int i = 0; i < 8; ++i) { b[sn][i] = lo[i]; b[sn][i + 8] = hi[i]; }
    }
#pragma unroll
    for (int sm = 0; sm < 2; ++sm)
#pragma unroll
      for (int sn = 0; sn < 2; ++sn)
        acc[sm][sn] = __builtin_amdgcn_wmma_f32_16x16x32_bf16(
            false, a[sm], false, b[sn], (short)0, acc[sm][sn], false, false);
  }
}

// ---------------- input projection: I_ext = X @ W_in^T + b ----------------
// block = 128 threads (4 waves as 1M x 4N), block tile = 32(M) x 128(N)
__global__ void k_gemm_in(const __bf16* __restrict__ X,
                          const __bf16* __restrict__ Win,
                          const float* __restrict__ bin,
                          float* __restrict__ Iext) {
  const int lane = threadIdx.x & 31;
  const int w    = threadIdx.x >> 5;           // 0..3 -> N slot
  const int Mbase = blockIdx.y * 32;
  const int Nbase = blockIdx.x * 128 + w * 32;

  v8f acc[2][2] = {};
  wmma_tile_gemm(X, DIN, Win, DIN, DIN, Mbase, Nbase, lane, acc);

  const int l15 = lane & 15;
  const int kh  = lane >> 4;
#pragma unroll
  for (int sm = 0; sm < 2; ++sm)
#pragma unroll
    for (int sn = 0; sn < 2; ++sn) {
      const int n = Nbase + sn * 16 + l15;
      const float bias = bin[n];
#pragma unroll
      for (int r = 0; r < 8; ++r) {
        const int m = Mbase + sm * 16 + r + 8 * kh;
        Iext[(size_t)m * NN + n] = acc[sm][sn][r] + bias;
      }
    }
}

// ---------------- one LIF time step ----------------
__global__ void k_step(const __bf16* __restrict__ s_in,
                       const __bf16* __restrict__ Weff,
                       const float* __restrict__ Iext,
                       float* __restrict__ V,
                       __bf16* __restrict__ s_out,
                       float* __restrict__ rate) {
  const int lane = threadIdx.x & 31;
  const int w    = threadIdx.x >> 5;           // 0..3 -> N slot
  const int Mbase = blockIdx.y * 32;
  const int Nbase = blockIdx.x * 128 + w * 32;

  v8f acc[2][2] = {};
  wmma_tile_gemm(s_in, NN, Weff, NN, NN, Mbase, Nbase, lane, acc);

  const int l15 = lane & 15;
  const int kh  = lane >> 4;
#pragma unroll
  for (int sm = 0; sm < 2; ++sm)
#pragma unroll
    for (int sn = 0; sn < 2; ++sn) {
      const int n = Nbase + sn * 16 + l15;
#pragma unroll
      for (int r = 0; r < 8; ++r) {
        const int m = Mbase + sm * 16 + r + 8 * kh;
        const size_t idx = (size_t)m * NN + n;
        float v = DECAY * V[idx] + Iext[idx] + acc[sm][sn][r];
        const float sp = (v >= VTH) ? 1.0f : 0.0f;
        V[idx] = v * (1.0f - sp);
        s_out[idx] = (__bf16)sp;
        rate[idx] += sp * (1.0f / (float)TSTEPS);
      }
    }
}

// ---------------- host launcher ----------------
extern "C" void kernel_launch(void* const* d_in, const int* in_sizes, int n_in,
                              void* d_out, int out_size, void* d_ws, size_t ws_size,
                              hipStream_t stream) {
  (void)in_sizes; (void)n_in; (void)out_size; (void)ws_size;
  const float* x    = (const float*)d_in[0];  // [256,1024]
  const float* Win  = (const float*)d_in[1];  // [2048,1024]
  const float* bin  = (const float*)d_in[2];  // [2048]
  const float* adj  = (const float*)d_in[3];  // [2048,2048]
  const float* mask = (const float*)d_in[4];  // [2048,2048]
  float* out = (float*)d_out;                 // [256,2048] firing rates

  char* ws = (char*)d_ws;
  __bf16* Weff = (__bf16*)(ws);                        //  8.0 MiB
  __bf16* Winb = (__bf16*)(ws + (size_t)(8u  << 20));  //  4.0 MiB
  __bf16* Xb   = (__bf16*)(ws + (size_t)(12u << 20));  //  0.5 MiB
  float*  Iext = (float*) (ws + (size_t)(13u << 20));  //  2.0 MiB
  float*  V    = (float*) (ws + (size_t)(15u << 20));  //  2.0 MiB
  __bf16* s0   = (__bf16*)(ws + (size_t)(17u << 20));  //  1.0 MiB
  __bf16* s1   = (__bf16*)(ws + (size_t)(18u << 20));  //  1.0 MiB (19 MiB total)

  k_maskmul_bf16<<<1024, 256, 0, stream>>>(adj, mask, Weff, NN * NN);
  k_cvt_bf16   <<<1024, 256, 0, stream>>>(Win, Winb, NN * DIN);
  k_cvt_bf16   <<<256,  256, 0, stream>>>(x, Xb, BATCH * DIN);
  k_zero_f32   <<<256,  256, 0, stream>>>(V, BATCH * NN);
  k_zero_f32   <<<256,  256, 0, stream>>>(out, BATCH * NN);
  k_zero_bf16  <<<256,  256, 0, stream>>>(s0, BATCH * NN);

  dim3 grid(NN / 128, BATCH / 32);  // (16, 8) = 128 workgroups
  dim3 block(128);                  // 4 waves
  k_gemm_in<<<grid, block, 0, stream>>>(Xb, Winb, bin, Iext);

  __bf16* scur = s0;
  __bf16* snxt = s1;
  for (int t = 0; t < TSTEPS; ++t) {
    k_step<<<grid, block, 0, stream>>>(scur, Weff, Iext, V, snxt, out);
    __bf16* tmp = scur; scur = snxt; snxt = tmp;
  }
}
